// EthicalRelationReasoning_64776696758655
// MI455X (gfx1250) — matrix operand order repaired and checked by reading, not compiled, not run on hardware
//
#include <hip/hip_runtime.h>

// ---------------------------------------------------------------------------
// EthicalRelationReasoning GNN forward for MI455X (gfx1250, wave32).
//
// Key algebraic folds (exact up to fp reassociation):
//   a_edge[e,h]  = edge_attr[e,:2] @ M2[2,4] + c2[4]          (real edges)
//   a_edge[n,h]  = (segsum(edge_attr,dst)[n]/max(cnt,1)) @ M2 + (cnt>0)*c2
//   a_src/a_dst  = h @ As/Ad  with As,Ad in [64,4]
// which removes the 850000x256 edge-feature GEMM (28 GFLOP + 870MB) entirely.
//
// xl = h @ lin_w runs on v_wmma_f32_16x16x32_bf16 (f32 accumulate).
// Aggregation / softmax use f32 global atomics (HBM-roofline bound: ~5 GB).
// ---------------------------------------------------------------------------

typedef __bf16 bf16_t;
typedef __attribute__((ext_vector_type(16))) __bf16 v16bf;
typedef __attribute__((ext_vector_type(8)))  __bf16 bf16x8;
typedef __attribute__((ext_vector_type(8)))  float  v8f;

#define NEG_SLOPE 0.2f
#define BN_EPS 1e-5f

// monotonic float<->uint mapping so atomicMax(uint) == float max
__device__ __forceinline__ unsigned enc_f32(float f) {
  unsigned u = __float_as_uint(f);
  return (u & 0x80000000u) ? ~u : (u | 0x80000000u);
}
__device__ __forceinline__ float dec_f32(unsigned u) {
  return __uint_as_float((u & 0x80000000u) ? (u & 0x7fffffffu) : ~u);
}
__device__ __forceinline__ unsigned short f2bf(float f) {
  unsigned u = __float_as_uint(f);
  u += 0x7fffu + ((u >> 16) & 1u);          // round-to-nearest-even
  return (unsigned short)(u >> 16);
}

// ------------------------------ utility fill -------------------------------
__global__ void k_fill_u32(unsigned* __restrict__ p, unsigned v, long n) {
  long i = (long)blockIdx.x * blockDim.x + threadIdx.x;
  if (i < n) p[i] = v;
}

// --------------------- per-node raw edge_attr segment sums -----------------
__global__ void k_edgestats(const float* __restrict__ eattr, const int* __restrict__ dst,
                            float* __restrict__ s2, float* __restrict__ cnt, int E) {
  int e = blockIdx.x * blockDim.x + threadIdx.x;
  if (e >= E) return;
  int d = dst[e];
  atomicAdd(&s2[d * 2 + 0], eattr[(long)e * 2 + 0]);
  atomicAdd(&s2[d * 2 + 1], eattr[(long)e * 2 + 1]);
  atomicAdd(&cnt[d], 1.0f);
}

// ------------------------------ node encoder -------------------------------
__global__ void k_nodeenc(const float* __restrict__ x, const float* __restrict__ W,
                          const float* __restrict__ b, float* __restrict__ h,
                          unsigned short* __restrict__ hbf, int n) {
  long i = (long)blockIdx.x * blockDim.x + threadIdx.x;
  if (i >= (long)n * 64) return;
  int node = (int)(i >> 6), c = (int)(i & 63);
  float a = b[c];
#pragma unroll
  for (int k = 0; k < 6; ++k) a += x[(long)node * 6 + k] * W[k * 64 + c];
  h[i] = a;
  hbf[i] = f2bf(a);
}

// ---------------- fold layer params into As, Ad [64,4], M2 [2,4], c2 [4] ----
__global__ void k_prep(const float* __restrict__ lin_w, const float* __restrict__ att_src,
                       const float* __restrict__ att_dst, const float* __restrict__ edge_w,
                       const float* __restrict__ att_edge, const float* __restrict__ We,
                       const float* __restrict__ be, float* __restrict__ As,
                       float* __restrict__ Ad, float* __restrict__ M2,
                       float* __restrict__ c2, int C) {
  __shared__ float W2[256];
  int d = threadIdx.x >> 2, h = threadIdx.x & 3;
  int HC = 4 * C;
  float as = 0.f, ad = 0.f, w2 = 0.f;
  for (int c = 0; c < C; ++c) {
    float lw = lin_w[d * HC + h * C + c];
    as += lw * att_src[h * C + c];
    ad += lw * att_dst[h * C + c];
    w2 += edge_w[d * HC + h * C + c] * att_edge[h * C + c];
  }
  As[d * 4 + h] = as;
  Ad[d * 4 + h] = ad;
  W2[d * 4 + h] = w2;
  __syncthreads();
  if (threadIdx.x < 8) {
    int i = threadIdx.x >> 2, hh = threadIdx.x & 3;
    float m = 0.f;
    for (int dd = 0; dd < 64; ++dd) m += We[i * 64 + dd] * W2[dd * 4 + hh];
    M2[i * 4 + hh] = m;
  }
  if (threadIdx.x < 4) {
    float cc = 0.f;
    for (int dd = 0; dd < 64; ++dd) cc += be[dd] * W2[dd * 4 + threadIdx.x];
    c2[threadIdx.x] = cc;
  }
}

// ------------- transpose + bf16-convert weights: [64,HC] -> [HC,64] --------
__global__ void k_wt_bf16(const float* __restrict__ W, unsigned short* __restrict__ Wt, int HC) {
  int i = blockIdx.x * blockDim.x + threadIdx.x;
  if (i >= HC * 64) return;
  int col = i >> 6, d = i & 63;
  Wt[col * 64 + d] = f2bf(W[d * HC + col]);
}

// ------------------------- WMMA GEMM: xl = h @ lin_w -----------------------
// A: bf16 [M,64] row-major.  B: bf16 [HC,64] (pre-transposed weights).
// Each wave computes one 16x16 f32 tile; K=64 -> 2 chained bf16 WMMAs.
__global__ void k_gemm_wmma(const unsigned short* __restrict__ Abits,
                            const unsigned short* __restrict__ Bbits,
                            float* __restrict__ Cmat, int M, int HC) {
  const int wave = threadIdx.x >> 5;
  const int lane = threadIdx.x & 31;
  const int tM = blockIdx.x * (blockDim.x >> 5) + wave;
  const int tN = blockIdx.y;
  if (tM * 16 >= M) return;            // wave-uniform: EXEC stays all-ones
  const int m0 = tM * 16, n0 = tN * 16;
  const bf16_t* A = (const bf16_t*)Abits;
  const bf16_t* B = (const bf16_t*)Bbits;
  const int rowA = m0 + (lane & 15);
  const int colB = n0 + (lane & 15);
  const int kA = (lane >> 4) << 3;     // A: +8 for upper half-lanes
  const int kB = (lane >> 4) << 3;     // B: +8 for upper half-lanes
  v8f acc = {};
#pragma unroll
  for (int k0 = 0; k0 < 64; k0 += 32) {
    bf16x8 alo = *(const bf16x8*)(A + (long)rowA * 64 + k0 + kA);        // K kb..kb+7
    bf16x8 ahi = *(const bf16x8*)(A + (long)rowA * 64 + k0 + kA + 16);   // K kb+16..kb+23
    bf16x8 blo = *(const bf16x8*)(B + (long)colB * 64 + k0 + kB);        // K kb..kb+7
    bf16x8 bhi = *(const bf16x8*)(B + (long)colB * 64 + k0 + kB + 16);   // K kb+16..kb+23
    v16bf a, b;
#pragma unroll
    for (int i = 0; i < 8; ++i) {
      a[i] = alo[i]; a[i + 8] = ahi[i];
      b[i] = blo[i]; b[i + 8] = bhi[i];
    }
    acc = __builtin_amdgcn_wmma_f32_16x16x32_bf16(false, a, false, b,
                                                  (short)0, acc, false, false);
  }
  const int rbase = m0 + ((lane >> 4) << 3);
  const int col = n0 + (lane & 15);
#pragma unroll
  for (int i = 0; i < 8; ++i) Cmat[(long)(rbase + i) * HC + col] = acc[i];
}

// ----------------------- a_src / a_dst = h @ As/Ad -------------------------
__global__ void k_nodeatt(const float* __restrict__ h, const float* __restrict__ As,
                          const float* __restrict__ Ad, float* __restrict__ asrc,
                          float* __restrict__ adst, int n) {
  int i = blockIdx.x * blockDim.x + threadIdx.x;
  int node = i >> 2, hh = i & 3;
  if (node >= n) return;
  float s = 0.f, d = 0.f;
#pragma unroll 8
  for (int k = 0; k < 64; ++k) {
    float hv = h[(long)node * 64 + k];
    s += hv * As[k * 4 + hh];
    d += hv * Ad[k * 4 + hh];
  }
  asrc[(long)node * 4 + hh] = s;
  adst[(long)node * 4 + hh] = d;
}

// ---------------- alpha = leaky_relu(...) + segment max over dst -----------
__global__ void k_alpha(const float* __restrict__ eattr, const float* __restrict__ s2,
                        const float* __restrict__ cnt, const float* __restrict__ M2,
                        const float* __restrict__ c2, const float* __restrict__ asrc,
                        const float* __restrict__ adst, const int* __restrict__ src,
                        const int* __restrict__ dst, float* __restrict__ alpha,
                        unsigned* __restrict__ menc, int E, int E2) {
  int e = blockIdx.x * blockDim.x + threadIdx.x;
  if (e >= E2) return;
  int s, d;
  float e0, e1, cb;
  if (e < E) {
    s = src[e]; d = dst[e];
    e0 = eattr[(long)e * 2]; e1 = eattr[(long)e * 2 + 1]; cb = 1.f;
  } else {                                  // self loop, fill_value='mean'
    int nn = e - E; s = d = nn;
    float c = cnt[nn];
    float inv = 1.f / fmaxf(c, 1.f);
    e0 = s2[(long)nn * 2] * inv; e1 = s2[(long)nn * 2 + 1] * inv;
    cb = (c > 0.f) ? 1.f : 0.f;
  }
#pragma unroll
  for (int h = 0; h < 4; ++h) {
    float a = asrc[(long)s * 4 + h] + adst[(long)d * 4 + h]
            + e0 * M2[h] + e1 * M2[4 + h] + cb * c2[h];
    a = (a > 0.f) ? a : NEG_SLOPE * a;
    alpha[(long)e * 4 + h] = a;
    atomicMax(&menc[(long)d * 4 + h], enc_f32(a));
  }
}

// ------------------ ex = exp(alpha - max); segment sum ---------------------
__global__ void k_expsum(float* __restrict__ alpha, const unsigned* __restrict__ menc,
                         float* __restrict__ den, const int* __restrict__ dst,
                         int E, int E2) {
  int e = blockIdx.x * blockDim.x + threadIdx.x;
  if (e >= E2) return;
  int d = (e < E) ? dst[e] : (e - E);
#pragma unroll
  for (int h = 0; h < 4; ++h) {
    float m = dec_f32(menc[(long)d * 4 + h]);
    float ex = __expf(alpha[(long)e * 4 + h] - m);
    alpha[(long)e * 4 + h] = ex;
    atomicAdd(&den[(long)d * 4 + h], ex);
  }
}

// ------------- agg[dst] += (ex/den) * xl[src]; one wave per edge -----------
template <int HC, int LOG2C>
__global__ void k_aggregate(const float* __restrict__ ex, const float* __restrict__ den,
                            const float* __restrict__ xl, const int* __restrict__ src,
                            const int* __restrict__ dst, float* __restrict__ agg,
                            int E, int E2) {
  int e = blockIdx.x * (blockDim.x >> 5) + (threadIdx.x >> 5);
  if (e >= E2) return;
  int lane = threadIdx.x & 31;
  int s, d;
  if (e < E) { s = src[e]; d = dst[e]; } else { s = d = e - E; }
  float w[4];
#pragma unroll
  for (int h = 0; h < 4; ++h)
    w[h] = ex[(long)e * 4 + h] / (den[(long)d * 4 + h] + 1e-16f);
  const int per = HC / 32;
  int ch0 = lane * per;
#pragma unroll
  for (int i = 0; i < per; ++i) {
    int ch = ch0 + i;
    float v = w[ch >> LOG2C] * xl[(long)s * HC + ch];
    atomicAdd(&agg[(long)d * HC + ch], v);
  }
}

// ------- head mean + bias, accumulate per-channel BN sum / sumsq -----------
template <int C>
__global__ void k_meanbias_stats(const float* __restrict__ agg, const float* __restrict__ bias,
                                 float* __restrict__ hm, float* __restrict__ sums, int n) {
  const int HC = C * 4;
  int c = threadIdx.x & (C - 1);
  int r = threadIdx.x / C;
  int R = blockDim.x / C;
  float s = 0.f, q = 0.f;
  for (long i = (long)blockIdx.x * R + r; i < n; i += (long)gridDim.x * R) {
    float a = 0.f;
#pragma unroll
    for (int h = 0; h < 4; ++h) a += agg[i * HC + h * C + c];
    a = a * 0.25f + bias[c];
    hm[i * C + c] = a;
    s += a; q += a * a;
  }
  __shared__ float ls[256], lq[256];
  ls[threadIdx.x] = s; lq[threadIdx.x] = q;
  __syncthreads();
  if (r == 0) {
    for (int rr = 1; rr < R; ++rr) { s += ls[rr * C + c]; q += lq[rr * C + c]; }
    atomicAdd(&sums[c], s);
    atomicAdd(&sums[C + c], q);
  }
}

// ------------- BN apply (+optional relu), emit bf16 for next GEMM ----------
template <int C>
__global__ void k_bnapply(float* __restrict__ hm, const float* __restrict__ sums,
                          const float* __restrict__ g, const float* __restrict__ b,
                          unsigned short* __restrict__ hbf, int n, int relu) {
  long i = (long)blockIdx.x * blockDim.x + threadIdx.x;
  if (i >= (long)n * C) return;
  int c = (int)(i & (C - 1));
  float mu = sums[c] / (float)n;
  float var = sums[C + c] / (float)n - mu * mu;
  float sc = g[c] * rsqrtf(var + BN_EPS);
  float sh = b[c] - mu * sc;
  float v = sc * hm[i] + sh;
  if (relu) v = fmaxf(v, 0.f);
  hm[i] = v;
  if (hbf) hbf[i] = f2bf(v);
}

// ------------------------------- pooling -----------------------------------
__global__ void k_pool(const float* __restrict__ ne, const int* __restrict__ batch,
                       float* __restrict__ gsum, unsigned* __restrict__ gmax,
                       float* __restrict__ cntb, int n) {
  long i = (long)blockIdx.x * blockDim.x + threadIdx.x;
  if (i >= (long)n * 32) return;
  int node = (int)(i >> 5), c = (int)(i & 31);
  int b = batch[node];
  float v = ne[i];
  atomicAdd(&gsum[b * 32 + c], v);
  atomicMax(&gmax[b * 32 + c], enc_f32(v));
  if (c == 0) atomicAdd(&cntb[b], 1.f);
}

// -------------------------------- MLP heads --------------------------------
__device__ __forceinline__ void mlp_head(const float* ge, const float* w1, const float* b1,
                                         const float* w2, const float* b2, int dout,
                                         float* out) {
  float hb[16];
#pragma unroll
  for (int j = 0; j < 16; ++j) {
    float a = b1[j];
    for (int i = 0; i < 32; ++i) a += ge[i] * w1[i * 16 + j];
    hb[j] = fmaxf(a, 0.f);
  }
  for (int o = 0; o < dout; ++o) {
    float a = b2[o];
#pragma unroll
    for (int j = 0; j < 16; ++j) a += hb[j] * w2[j * dout + o];
    out[o] = a;
  }
}

__global__ void k_heads(const float* __restrict__ gsum, const unsigned* __restrict__ gmax,
                        const float* __restrict__ cntb,
                        const float* eb1, const float* eb2, const float* ew1, const float* ew2,
                        const float* mb1, const float* mb2, const float* mw1, const float* mw2,
                        const float* ob1, const float* ob2, const float* ow1, const float* ow2,
                        float* __restrict__ ge_out, float* __restrict__ eth_out,
                        float* __restrict__ man_out, float* __restrict__ mor_out) {
  int b = threadIdx.x;
  if (b >= 64) return;
  float ge[32];
  float c = fmaxf(cntb[b], 1.f);
  for (int i = 0; i < 32; ++i) {
    float s = gsum[b * 32 + i];
    float mx = dec_f32(gmax[b * 32 + i]);
    ge[i] = (s / c + mx + s) * (1.f / 3.f);
    ge_out[b * 32 + i] = ge[i];
  }
  float o1, o6[6];
  mlp_head(ge, ew1, eb1, ew2, eb2, 1, &o1);
  eth_out[b] = 1.f / (1.f + __expf(-o1));
  mlp_head(ge, mw1, mb1, mw2, mb2, 1, &o1);
  man_out[b] = 1.f / (1.f + __expf(-o1));
  mlp_head(ge, ow1, ob1, ow2, ob2, 6, o6);
  for (int o = 0; o < 6; ++o) mor_out[b * 6 + o] = o6[o];
}

// ===========================================================================
extern "C" void kernel_launch(void* const* d_in, const int* in_sizes, int n_in,
                              void* d_out, int out_size, void* d_ws, size_t ws_size,
                              hipStream_t stream) {
  (void)in_sizes; (void)n_in; (void)out_size; (void)ws_size;
  const int N = 50000, E = 800000, E2 = E + N;

  // --- input unpack: dict order; params pytree flattened in sorted-key order
  const float* x         = (const float*)d_in[0];   // [N,6]
  const float* edge_attr = (const float*)d_in[1];   // [E,2]
  const float* ee_b      = (const float*)d_in[2];   // edge_enc.b [64]
  const float* ee_w      = (const float*)d_in[3];   // edge_enc.w [2,64]
  const float* eth_b1    = (const float*)d_in[4];
  const float* eth_b2    = (const float*)d_in[5];
  const float* eth_w1    = (const float*)d_in[6];
  const float* eth_w2    = (const float*)d_in[7];
  // layers: base 8 + 8*li : att_dst, att_edge, att_src, bias, bn_b, bn_g, edge_w, lin_w
  const float* man_b1    = (const float*)d_in[32];
  const float* man_b2    = (const float*)d_in[33];
  const float* man_w1    = (const float*)d_in[34];
  const float* man_w2    = (const float*)d_in[35];
  const float* mor_b1    = (const float*)d_in[36];
  const float* mor_b2    = (const float*)d_in[37];
  const float* mor_w1    = (const float*)d_in[38];
  const float* mor_w2    = (const float*)d_in[39];
  const float* ne_b      = (const float*)d_in[40];  // node_enc.b [64]
  const float* ne_w      = (const float*)d_in[41];  // node_enc.w [6,64]
  const int*   eidx      = (const int*)d_in[42];    // [2,E]
  const int*   batch     = (const int*)d_in[43];    // [N]
  const int* srcp = eidx;
  const int* dstp = eidx + E;

  // --- output layout: node_emb | graph_emb | ethics | manip | moral
  float* out       = (float*)d_out;
  float* out_node  = out;                  // [N,32]
  float* out_ge    = out + (long)N * 32;   // [64,32]
  float* out_eth   = out_ge + 64 * 32;     // [64]
  float* out_man   = out_eth + 64;         // [64]
  float* out_mor   = out_man + 64;         // [64,6]

  // --- workspace carve-up (256B aligned)
  char* base = (char*)d_ws;
  size_t off = 0;
  auto carve = [&](size_t bytes) -> void* {
    void* p = base + off;
    off += (bytes + 255) & ~(size_t)255;
    return p;
  };
  float*          h     = (float*)carve((size_t)N * 64 * 4);
  unsigned short* hbf   = (unsigned short*)carve((size_t)N * 64 * 2);
  float*          xl    = (float*)carve((size_t)N * 256 * 4);
  float*          agg   = (float*)carve((size_t)N * 256 * 4);
  float*          asrc  = (float*)carve((size_t)N * 4 * 4);
  float*          adst  = (float*)carve((size_t)N * 4 * 4);
  float*          alpha = (float*)carve((size_t)E2 * 4 * 4);
  unsigned*       menc  = (unsigned*)carve((size_t)N * 4 * 4);
  float*          den   = (float*)carve((size_t)N * 4 * 4);
  float*          s2    = (float*)carve((size_t)N * 2 * 4);
  float*          cnt   = (float*)carve((size_t)N * 4);
  unsigned short* wtbf  = (unsigned short*)carve((size_t)256 * 64 * 2);
  float*          As    = (float*)carve(256 * 4);
  float*          Ad    = (float*)carve(256 * 4);
  float*          M2    = (float*)carve(8 * 4);
  float*          c2    = (float*)carve(4 * 4);
  float*          sums  = (float*)carve(128 * 4);
  float*          gsum  = (float*)carve(64 * 32 * 4);
  unsigned*       gmax  = (unsigned*)carve(64 * 32 * 4);
  float*          cntb  = (float*)carve(64 * 4);

  auto blks = [](long n, int bs) { return (unsigned)((n + bs - 1) / bs); };
  const unsigned NEG_INF_ENC = 0x007FFFFFu;  // enc(-inf)

  // --- graph-invariant precompute (re-done every call: deterministic)
  k_fill_u32<<<blks((long)N * 3, 256), 256, 0, stream>>>((unsigned*)s2, 0u, (long)N * 3);
  k_edgestats<<<blks(E, 256), 256, 0, stream>>>(edge_attr, dstp, s2, cnt, E);
  k_nodeenc<<<blks((long)N * 64, 256), 256, 0, stream>>>(x, ne_w, ne_b, h, hbf, N);

  // --- GAT layers
  for (int li = 0; li < 3; ++li) {
    const int C = (li < 2) ? 64 : 32;
    const int HC = 4 * C;
    const float** L = (const float**)(d_in + 8 + 8 * li);
    const float* att_dst  = L[0];
    const float* att_edge = L[1];
    const float* att_src  = L[2];
    const float* bias     = L[3];
    const float* bn_b     = L[4];
    const float* bn_g     = L[5];
    const float* edge_w   = L[6];
    const float* lin_w    = L[7];

    k_prep<<<1, 256, 0, stream>>>(lin_w, att_src, att_dst, edge_w, att_edge,
                                  ee_w, ee_b, As, Ad, M2, c2, C);
    k_wt_bf16<<<blks(HC * 64, 256), 256, 0, stream>>>(lin_w, wtbf, HC);

    dim3 ggrid(blks(N / 16, 8), HC / 16);
    k_gemm_wmma<<<ggrid, 256, 0, stream>>>(hbf, wtbf, xl, N, HC);

    k_nodeatt<<<blks((long)N * 4, 256), 256, 0, stream>>>(h, As, Ad, asrc, adst, N);

    k_fill_u32<<<blks((long)N * 4, 256), 256, 0, stream>>>(menc, NEG_INF_ENC, (long)N * 4);
    k_fill_u32<<<blks((long)N * 4, 256), 256, 0, stream>>>((unsigned*)den, 0u, (long)N * 4);
    k_fill_u32<<<blks((long)N * HC, 256), 256, 0, stream>>>((unsigned*)agg, 0u, (long)N * HC);

    k_alpha<<<blks(E2, 256), 256, 0, stream>>>(edge_attr, s2, cnt, M2, c2, asrc, adst,
                                               srcp, dstp, alpha, menc, E, E2);
    k_expsum<<<blks(E2, 256), 256, 0, stream>>>(alpha, menc, den, dstp, E, E2);

    if (li < 2)
      k_aggregate<256, 6><<<blks(E2, 8), 256, 0, stream>>>(alpha, den, xl, srcp, dstp,
                                                           agg, E, E2);
    else
      k_aggregate<128, 5><<<blks(E2, 8), 256, 0, stream>>>(alpha, den, xl, srcp, dstp,
                                                           agg, E, E2);

    float* hm = (li < 2) ? h : out_node;
    k_fill_u32<<<1, 256, 0, stream>>>((unsigned*)sums, 0u, 2 * C);
    if (li < 2) {
      k_meanbias_stats<64><<<512, 256, 0, stream>>>(agg, bias, hm, sums, N);
      k_bnapply<64><<<blks((long)N * 64, 256), 256, 0, stream>>>(hm, sums, bn_g, bn_b,
                                                                 hbf, N, 1);
    } else {
      k_meanbias_stats<32><<<512, 256, 0, stream>>>(agg, bias, hm, sums, N);
      k_bnapply<32><<<blks((long)N * 32, 256), 256, 0, stream>>>(hm, sums, bn_g, bn_b,
                                                                 nullptr, N, 0);
    }
  }

  // --- pooling + heads
  k_fill_u32<<<blks(64 * 32, 256), 256, 0, stream>>>((unsigned*)gsum, 0u, 64 * 32);
  k_fill_u32<<<blks(64 * 32, 256), 256, 0, stream>>>(gmax, NEG_INF_ENC, 64 * 32);
  k_fill_u32<<<1, 64, 0, stream>>>((unsigned*)cntb, 0u, 64);
  k_pool<<<blks((long)N * 32, 256), 256, 0, stream>>>(out_node, batch, gsum, gmax, cntb, N);
  k_heads<<<1, 64, 0, stream>>>(gsum, gmax, cntb,
                                eth_b1, eth_b2, eth_w1, eth_w2,
                                man_b1, man_b2, man_w1, man_w2,
                                mor_b1, mor_b2, mor_w1, mor_w2,
                                out_ge, out_eth, out_man, out_mor);
}
